// LLANetHead_71597104824791
// MI455X (gfx1250) — matrix operand order repaired and checked by reading, not compiled, not run on hardware
//
#include <hip/hip_runtime.h>
#include <hip/hip_bf16.h>
#include <math.h>

#define NB 32          // B
#define NC 64          // C
#define NMID 48        // MID
#define NP 192         // P
#define NS 36          // S
#define NOFF 72        // N_OFFSETS
#define NROI (NB*NP)   // 6144
#define KFC (NC*NS)    // 2304

typedef __attribute__((ext_vector_type(16))) _Float16 v16h;
typedef __attribute__((ext_vector_type(8)))  float    v8f;

// ---------------------------------------------------------------------------
// WMMA fragment loaders (wave32 f16 16x16x32; A and B share this file's contiguous k order)
// A (16x32, MxK): lane m=lane&15, K = kh..kh+15, kh=16*(lane>=16)
// B (32x16, KxN) loaded from transposed storage Bt[n][k]:
//   lane n=lane&15, K = kb..kb+15, kb=16*(lane>=16)  (16 contiguous halves)
// C/D (16x16 f32): row = v + 8*(lane>=16), col = lane&15
// ---------------------------------------------------------------------------
__device__ __forceinline__ v16h wmma_load_a(const _Float16* A, int lda, int m0, int k0) {
  const int lane = threadIdx.x & 31;
  const _Float16* p = A + (size_t)(m0 + (lane & 15)) * lda + k0 + ((lane >> 4) << 4);
  v16h a;
#pragma unroll
  for (int j = 0; j < 8; ++j) a[j] = p[j];
#pragma unroll
  for (int j = 0; j < 8; ++j) a[8 + j] = p[8 + j];
  return a;
}

__device__ __forceinline__ v16h wmma_load_bt(const _Float16* Bt, int ldb, int n0, int k0) {
  const int lane = threadIdx.x & 31;
  const _Float16* p = Bt + (size_t)(n0 + (lane & 15)) * ldb + k0 + ((lane >> 4) << 4);
  v16h b;
#pragma unroll
  for (int j = 0; j < 16; ++j) b[j] = p[j];
  return b;
}

// Bt load with per-row (per-lane) range guard -> zero rows outside [0, rlimit)
__device__ __forceinline__ v16h wmma_load_bt_guard(const _Float16* Bt, int ldb, int rowbase,
                                                   int k0, int rlimit) {
  const int lane = threadIdx.x & 31;
  const int r = rowbase + (lane & 15);
  v16h b;
  if (r >= 0 && r < rlimit) {
    const _Float16* p = Bt + (size_t)r * ldb + k0 + ((lane >> 4) << 4);
#pragma unroll
    for (int j = 0; j < 16; ++j) b[j] = p[j];
  } else {
#pragma unroll
    for (int j = 0; j < 16; ++j) b[j] = (_Float16)0.f;
  }
  return b;
}

__device__ __forceinline__ v8f wmma_mac(v16h a, v16h b, v8f c) {
  return __builtin_amdgcn_wmma_f32_16x16x32_f16(false, a, false, b, (short)0, c, false, false);
}

// ---------------------------------------------------------------------------
// Generic batched WMMA GEMM: out = epilogue( A(MxK) * Bt(NxK)^T )
// 4 waves per block along M (M multiple of 64). Each wave computes a
// 16 x (NSUB*16) strip; NSUB is a compile-time constant so the accumulators
// stay in fixed registers (one A-fragment load feeds NSUB WMMAs).
// K multiple of 32; Bt rows padded to 16. nbase = column offset of strip 0.
// ---------------------------------------------------------------------------
template <int NSUB>
__global__ __launch_bounds__(128)
void gemm16(const _Float16* __restrict__ A, int lda, long aBS,
            const _Float16* __restrict__ Bt, int ldb, long bBS,
            float* __restrict__ O32, _Float16* __restrict__ O16, int ldc, long oBS,
            int M, int N, int K, int nbase,
            const float* __restrict__ colscale, const float* __restrict__ colbias,
            const float* __restrict__ rowscale, const float* __restrict__ rowbias,
            const float* __restrict__ addsrc, long addBS, int addld,
            float alpha, int relu, int trans, int ffperm) {
  const int z   = blockIdx.z;
  const int wid = threadIdx.x >> 5;
  const int m0  = blockIdx.x * 64 + wid * 16;
  const int n00 = nbase + blockIdx.y * (NSUB * 16);
  const _Float16* Ab = A + (size_t)z * aBS;
  const _Float16* Bb = Bt + (size_t)z * bBS;
  v8f acc[NSUB];
#pragma unroll
  for (int t = 0; t < NSUB; ++t)
#pragma unroll
    for (int q = 0; q < 8; ++q) acc[t][q] = 0.f;
  for (int k = 0; k < K; k += 32) {
    v16h a = wmma_load_a(Ab, lda, m0, k);
#pragma unroll
    for (int t = 0; t < NSUB; ++t) {
      v16h b = wmma_load_bt(Bb, ldb, n00 + t * 16, k);
      acc[t] = wmma_mac(a, b, acc[t]);
    }
  }
  const int lane = threadIdx.x & 31;
  const int hi = lane >> 4;
#pragma unroll
  for (int t = 0; t < NSUB; ++t) {
    const int n = n00 + t * 16 + (lane & 15);
#pragma unroll
    for (int v = 0; v < 8; ++v) {
      const int m = m0 + hi * 8 + v;
      if (m >= M || n >= N) continue;
      float val = acc[t][v] * alpha;
      if (rowscale) val = val * rowscale[m];
      if (rowbias)  val = val + rowbias[m];
      if (colscale) val = val * colscale[n];
      if (colbias)  val = val + colbias[n];
      if (addsrc)   val = val + addsrc[(size_t)z * addBS + (size_t)m * addld + n];
      if (relu)     val = fmaxf(val, 0.f);
      if (ffperm) {
        long r = (long)z * NP + m;             // roi flat row (b*P+p)
        long row = (r & 31) * NP + (r >> 5);   // ff permutation
        O16[row * ldc + n] = (_Float16)val;
        continue;
      }
      long row = m, col = n;
      if (trans) { long tt = row; row = col; col = tt; }
      if (O16) O16[(size_t)z * oBS + row * ldc + col] = (_Float16)val;
      else     O32[(size_t)z * oBS + row * ldc + col] = val;
    }
  }
}

// ---------------------------------------------------------------------------
// Weight packing / conversion kernels
// ---------------------------------------------------------------------------
__global__ void pack_rgw(const float* __restrict__ w, _Float16* __restrict__ dst) {
  // src (3,48,64,9) -> dst [s][k][co][ci]
  int i = blockIdx.x * blockDim.x + threadIdx.x;
  if (i >= 3 * 9 * 48 * 64) return;
  int ci = i & 63; int t = i >> 6;
  int co = t % 48; int sk = t / 48;
  int k = sk % 9;  int s = sk / 9;
  dst[i] = (_Float16)w[(((s * 48 + co) * 64 + ci) * 9) + k];
}

__global__ void pack_catw(const float* __restrict__ w, _Float16* __restrict__ dst,
                          int Cin, int CH) {
  // src (64,Cin,9) -> dst [k][co][ci_pad(CH)] with zero pad
  int i = blockIdx.x * blockDim.x + threadIdx.x;
  if (i >= 9 * 64 * CH) return;
  int ci = i % CH; int t = i / CH;
  int co = t % 64; int k = t / 64;
  dst[i] = (ci < Cin) ? (_Float16)w[((size_t)co * Cin + ci) * 9 + k] : (_Float16)0.f;
}

__global__ void cvt_pad_f16(const float* __restrict__ src, _Float16* __restrict__ dst,
                            int rows, int cols, int rowspad) {
  int i = blockIdx.x * blockDim.x + threadIdx.x;
  if (i >= rowspad * cols) return;
  int r = i / cols;
  dst[i] = (r < rows) ? (_Float16)src[i] : (_Float16)0.f;
}

__global__ void proto_norm(const float* __restrict__ proto, _Float16* __restrict__ dst) {
  __shared__ float sb[64];
  int r = blockIdx.x, c = threadIdx.x;
  float v = (r < 15) ? proto[r * 64 + c] : 0.f;
  sb[c] = v * v; __syncthreads();
  for (int st = 32; st > 0; st >>= 1) { if (c < st) sb[c] += sb[c + st]; __syncthreads(); }
  float d = fmaxf(sqrtf(sb[0]), 1e-12f);
  dst[r * 64 + c] = (r < 15) ? (_Float16)(v / d) : (_Float16)0.f;
}

// ---------------------------------------------------------------------------
// Grid-sample pooling: pooledT[n][t][c]  (t-major for LDS staging in conv)
// ---------------------------------------------------------------------------
__global__ __launch_bounds__(256)
void pool_kernel(const float* __restrict__ feat, int h, int w,
                 const float* __restrict__ lines, int stage,
                 _Float16* __restrict__ out) {
  const int n = blockIdx.x;
  const int b = n / NP, p = n % NP;
  for (int e = threadIdx.x; e < NS * NC; e += 256) {
    const int c = e & 63, si = e >> 6;
    const int sxr = (int)floorf((float)(35 - si) * (71.f / 35.f)); // SX[35-si]
    const float gy = (1.f - (float)sxr * (1.f / 71.f)) * 2.f - 1.f;
    const float px = (stage == 0) ? ((float)p + 0.5f) / (float)NP
                                  : lines[(size_t)n * 78 + 6 + sxr];
    const float gx = px * 2.f - 1.f;
    const float ix = (gx + 1.f) * 0.5f * (float)(w - 1);
    const float iy = (gy + 1.f) * 0.5f * (float)(h - 1);
    const float x0 = floorf(ix), y0 = floorf(iy);
    const float* f = feat + ((size_t)b * NC + c) * h * w;
    float acc = 0.f;
#pragma unroll
    for (int dy = 0; dy < 2; ++dy)
#pragma unroll
      for (int dx = 0; dx < 2; ++dx) {
        float xc = x0 + dx, yc = y0 + dy;
        float wt = (dx ? (ix - x0) : (x0 + 1.f - ix)) * (dy ? (iy - y0) : (y0 + 1.f - iy));
        if (xc >= 0.f && xc <= (float)(w - 1) && yc >= 0.f && yc <= (float)(h - 1)) {
          int xi = (int)xc, yi = (int)yc;
          acc += f[yi * w + xi] * wt;
        }
      }
    out[((size_t)n * NS + si) * NC + c] = (_Float16)acc;
  }
}

// ---------------------------------------------------------------------------
// Fused conv stack per ROI: rg_conv(s) for i<=stage, concat, catconv -> rcflat
// One workgroup (4 waves) per ROI. All matmuls via WMMA over shifted K-chunks.
// ---------------------------------------------------------------------------
template <int STAGE>
__global__ __launch_bounds__(128)
void conv_stack(const _Float16* __restrict__ pooled0,
                const _Float16* __restrict__ pooled1,
                const _Float16* __restrict__ pooled2,
                const _Float16* __restrict__ rgw,      // [3][9][48][64]
                const float* __restrict__ rg_scale,    // (3,48)
                const float* __restrict__ rg_bias,
                const _Float16* __restrict__ catw,     // [9][64][CH]
                const float* __restrict__ cat_scale,   // (64) stage row
                const float* __restrict__ cat_bias,
                _Float16* __restrict__ rcflat) {       // [n][c*36+t]
  constexpr int CH = (STAGE == 0) ? 64 : (STAGE == 1) ? 96 : 160;
  __shared__ __align__(16) _Float16 xT[NS * NC];       // pooledT tile [t][ci]
  __shared__ __align__(16) _Float16 fT[NS * CH];       // feats [t][ch], zero-padded
  const int n = blockIdx.x;
  const int tid = threadIdx.x;
  const int wid = tid >> 5;
  const int lane = tid & 31;
  const int hi = lane >> 4;

  for (int i = tid; i < NS * CH; i += 128) fT[i] = (_Float16)0.f;

  for (int ist = 0; ist <= STAGE; ++ist) {
    __syncthreads();
    const _Float16* src = (ist == 0) ? pooled0 : (ist == 1) ? pooled1 : pooled2;
    src += (size_t)n * NS * NC;
    for (int i = tid; i < NS * NC / 8; i += 128)
      ((int4*)xT)[i] = ((const int4*)src)[i];
    __syncthreads();
    // rg conv ist: Cout=48 (3 M tiles) x 3 N tiles
    for (int t9 = wid; t9 < 9; t9 += 4) {
      const int mt = t9 / 3, nt = t9 % 3;
      const int tcol = nt * 16 + (lane & 15);
      v8f acc;
#pragma unroll
      for (int q = 0; q < 8; ++q) acc[q] = 0.f;
      for (int k = 0; k < 9; ++k) {
#pragma unroll
        for (int cc = 0; cc < 2; ++cc) {
          v16h a = wmma_load_a(rgw + (size_t)((ist * 9 + k) * 48) * 64, 64, mt * 16, cc * 32);
          v16h b = wmma_load_bt_guard(xT, NC, nt * 16 + k - 4, cc * 32, NS);
          acc = wmma_mac(a, b, acc);
        }
      }
      if (tcol < NS) {
#pragma unroll
        for (int v = 0; v < 8; ++v) {
          int co = mt * 16 + hi * 8 + v;
          float val = acc[v] * rg_scale[ist * 48 + co] + rg_bias[ist * 48 + co];
          fT[tcol * CH + ist * 48 + co] = (_Float16)fmaxf(val, 0.f);
        }
      }
    }
  }
  __syncthreads();
  // catconv: Cout=64 (4 M tiles) x 3 N tiles, K = 9 shifts x CH
  for (int t12 = wid; t12 < 12; t12 += 4) {
    const int mt = t12 / 3, nt = t12 % 3;
    const int tcol = nt * 16 + (lane & 15);
    v8f acc;
#pragma unroll
    for (int q = 0; q < 8; ++q) acc[q] = 0.f;
    for (int k = 0; k < 9; ++k) {
#pragma unroll
      for (int cc = 0; cc < CH / 32; ++cc) {
        v16h a = wmma_load_a(catw + (size_t)k * 64 * CH, CH, mt * 16, cc * 32);
        v16h b = wmma_load_bt_guard(fT, CH, nt * 16 + k - 4, cc * 32, NS);
        acc = wmma_mac(a, b, acc);
      }
    }
    if (tcol < NS) {
#pragma unroll
      for (int v = 0; v < 8; ++v) {
        int co = mt * 16 + hi * 8 + v;
        float val = acc[v] * cat_scale[co] + cat_bias[co];
        rcflat[(size_t)n * KFC + co * NS + tcol] = (_Float16)fmaxf(val, 0.f);
      }
    }
  }
}

// ---------------------------------------------------------------------------
// Elementwise / reduction glue
// ---------------------------------------------------------------------------
__global__ __launch_bounds__(64)
void ln_relu(const float* __restrict__ x, float* __restrict__ y,
             const float* __restrict__ gamma, const float* __restrict__ beta) {
  __shared__ float sb[64];
  const int r = blockIdx.x, c = threadIdx.x;
  const float v = x[(size_t)r * 64 + c];
  sb[c] = v; __syncthreads();
  for (int st = 32; st > 0; st >>= 1) { if (c < st) sb[c] += sb[c + st]; __syncthreads(); }
  const float mu = sb[0] * (1.f / 64.f);
  __syncthreads();
  const float d = v - mu;
  sb[c] = d * d; __syncthreads();
  for (int st = 32; st > 0; st >>= 1) { if (c < st) sb[c] += sb[c + st]; __syncthreads(); }
  const float var = sb[0] * (1.f / 64.f);
  float out = d * rsqrtf(var + 1e-5f) * gamma[c] + beta[c];
  y[(size_t)r * 64 + c] = fmaxf(out, 0.f);
}

__global__ void resize_kernel(const float* __restrict__ feat, int h, int w,
                              _Float16* __restrict__ out) {   // (B,256,64)
  size_t i = (size_t)blockIdx.x * 256 + threadIdx.x;
  if (i >= (size_t)NB * 256 * 64) return;
  int c = (int)(i & 63); long t = i >> 6;
  int pix = (int)(t % 256); int b = (int)(t / 256);
  if (pix >= 250) { out[i] = (_Float16)0.f; return; }
  int rr = pix / 25, ww = pix % 25;
  int hi = rr * h / 10, wi = ww * w / 25;
  out[i] = (_Float16)feat[(((size_t)b * NC + c) * h + hi) * w + wi];
}

__global__ void query_kernel(const float* __restrict__ roi, const float* __restrict__ fqw,
                             const float* __restrict__ fqb, _Float16* __restrict__ q) {
  size_t i = (size_t)blockIdx.x * 256 + threadIdx.x;
  if (i >= (size_t)NROI * 64) return;
  int p = (int)((i >> 6) % NP);
  float v = roi[i] * fqw[p] + fqb[p];
  q[i] = (_Float16)fmaxf(v, 0.f);
}

__global__ __launch_bounds__(256)
void softmax250(const float* __restrict__ sim, _Float16* __restrict__ simh) {
  __shared__ float sb[256];
  const int r = blockIdx.x, c = threadIdx.x;
  const float* row = sim + (size_t)r * 256;
  float v = (c < 250) ? row[c] * 0.125f : -1e30f;   // * C^-0.5
  sb[c] = v; __syncthreads();
  for (int st = 128; st > 0; st >>= 1) { if (c < st) sb[c] = fmaxf(sb[c], sb[c + st]); __syncthreads(); }
  const float mx = sb[0]; __syncthreads();
  float e = (c < 250) ? expf(v - mx) : 0.f;
  sb[c] = e; __syncthreads();
  for (int st = 128; st > 0; st >>= 1) { if (c < st) sb[c] += sb[c + st]; __syncthreads(); }
  simh[(size_t)r * 256 + c] = (_Float16)(e / sb[0]);
}

__global__ __launch_bounds__(64)
void rownorm64(const float* __restrict__ x, _Float16* __restrict__ y) {
  __shared__ float sb[64];
  const int r = blockIdx.x, c = threadIdx.x;
  const float v = x[(size_t)r * 64 + c];
  sb[c] = v * v; __syncthreads();
  for (int st = 32; st > 0; st >>= 1) { if (c < st) sb[c] += sb[c + st]; __syncthreads(); }
  const float d = fmaxf(sqrtf(sb[0]), 1e-12f);
  y[(size_t)r * 64 + c] = (_Float16)(v / d);
}

__global__ __launch_bounds__(128)
void lines_kernel(const float* __restrict__ clsl, const float* __restrict__ regb,
                  float* __restrict__ lines, float* __restrict__ pred, int stage) {
  const int r = blockIdx.x;
  float pri2, pri3, pri4;
  if (stage == 0) { pri2 = 0.5f; pri3 = 0.1f; pri4 = 0.0f; }
  else {
    pri2 = lines[(size_t)r * 78 + 2];
    pri3 = lines[(size_t)r * 78 + 3];
    pri4 = lines[(size_t)r * 78 + 4];
  }
  __syncthreads();
  const float* rg = regb + (size_t)r * 76;
  const float p2 = pri2 + rg[0];
  const float p3 = pri3 + rg[1];
  const float p4 = pri4 + rg[2];
  const float p5 = rg[3];
  const float invtan = 1.f / tanf(p4 * 3.14159265358979323846f + 1e-5f);
  float* L = lines + (size_t)r * 78;
  float* D = pred + (size_t)r * 78;
  const int j = threadIdx.x;
  if (j < NOFF) {
    const float prior_y = 1.f - (float)j * (1.f / 71.f);
    const float geom = (p3 * 799.f + (1.f - prior_y - p2) * 320.f * invtan) * (1.f / 799.f);
    L[6 + j] = geom;
    D[6 + j] = geom + rg[4 + j];
  }
  if (j == 72) {
    const float c0 = clsl[(size_t)r * 2], c1 = clsl[(size_t)r * 2 + 1];
    L[0] = c0; L[1] = c1; L[2] = p2; L[3] = p3; L[4] = p4; L[5] = p5;
    D[0] = c0; D[1] = c1; D[2] = p2; D[3] = p3; D[4] = p4; D[5] = p5;
  }
}

// ---------------------------------------------------------------------------
// Host orchestration
// ---------------------------------------------------------------------------
extern "C" void kernel_launch(void* const* d_in, const int* in_sizes, int n_in,
                              void* d_out, int out_size, void* d_ws, size_t ws_size,
                              hipStream_t stream) {
  (void)in_sizes; (void)n_in; (void)out_size; (void)ws_size;
  const float* feat_s8   = (const float*)d_in[0];
  const float* feat_s16  = (const float*)d_in[1];
  const float* feat_s32  = (const float*)d_in[2];
  const float* rg_conv_w = (const float*)d_in[3];
  const float* rg_scale  = (const float*)d_in[4];
  const float* rg_bias   = (const float*)d_in[5];
  const float* catw0     = (const float*)d_in[6];
  const float* catw1     = (const float*)d_in[7];
  const float* catw2     = (const float*)d_in[8];
  const float* cat_scale = (const float*)d_in[9];
  const float* cat_bias  = (const float*)d_in[10];
  const float* fc_w      = (const float*)d_in[11];
  const float* fc_b      = (const float*)d_in[12];
  const float* ln_gamma  = (const float*)d_in[13];
  const float* ln_beta   = (const float*)d_in[14];
  const float* fkey_w    = (const float*)d_in[15];
  const float* fkey_s    = (const float*)d_in[16];
  const float* fkey_b    = (const float*)d_in[17];
  const float* fval_w    = (const float*)d_in[18];
  const float* fval_b    = (const float*)d_in[19];
  const float* fq_w      = (const float*)d_in[20];
  const float* fq_b      = (const float*)d_in[21];
  const float* attn_s    = (const float*)d_in[22];
  const float* attn_b    = (const float*)d_in[23];
  const float* cls_w1    = (const float*)d_in[24];
  const float* cls_b1    = (const float*)d_in[25];
  const float* cls_w2    = (const float*)d_in[26];
  const float* cls_b2    = (const float*)d_in[27];
  const float* reg_w1    = (const float*)d_in[28];
  const float* reg_b1    = (const float*)d_in[29];
  const float* reg_w2    = (const float*)d_in[30];
  const float* reg_b2    = (const float*)d_in[31];
  const float* clsh_w    = (const float*)d_in[32];
  const float* clsh_b    = (const float*)d_in[33];
  const float* regh_w    = (const float*)d_in[34];
  const float* regh_b    = (const float*)d_in[35];
  const float* cat_w1    = (const float*)d_in[36];
  const float* cat_b1    = (const float*)d_in[37];
  const float* cat_w2    = (const float*)d_in[38];
  const float* cat_b2    = (const float*)d_in[39];
  const float* protos    = (const float*)d_in[40];
  const float* attr_w1   = (const float*)d_in[41];
  const float* attr_b1   = (const float*)d_in[42];
  const float* attr_w2   = (const float*)d_in[43];
  const float* attr_b2   = (const float*)d_in[44];
  const float* attrh_w   = (const float*)d_in[45];
  const float* attrh_b   = (const float*)d_in[46];
  float* dout = (float*)d_out;

  size_t off = 0;
  auto alloc = [&](size_t bytes) -> void* {
    off = (off + 255) & ~(size_t)255;
    void* p = (char*)d_ws + off;
    off += bytes;
    return p;
  };
  typedef _Float16 h16;
  h16* rgw16     = (h16*)alloc((size_t)3 * 9 * 48 * 64 * 2);
  h16* catw16    = (h16*)alloc((size_t)(9 * 64 * 64 + 9 * 64 * 96 + 9 * 64 * 160) * 2);
  h16* fcw16     = (h16*)alloc((size_t)64 * KFC * 2);
  h16* fkeyw16   = (h16*)alloc(4096 * 2);
  h16* fvalw16   = (h16*)alloc(4096 * 2);
  h16* clsw1_16  = (h16*)alloc(4096 * 2);
  h16* clsw2_16  = (h16*)alloc(4096 * 2);
  h16* regw1_16  = (h16*)alloc(4096 * 2);
  h16* regw2_16  = (h16*)alloc(4096 * 2);
  h16* catw1_16  = (h16*)alloc(4096 * 2);
  h16* catw2_16  = (h16*)alloc(4096 * 2);
  h16* attrw1_16 = (h16*)alloc(4096 * 2);
  h16* attrw2_16 = (h16*)alloc(4096 * 2);
  h16* clshead16 = (h16*)alloc((size_t)16 * 64 * 2);
  h16* reghead16 = (h16*)alloc((size_t)80 * 64 * 2);
  h16* attrhead16= (h16*)alloc((size_t)16 * 64 * 2);
  h16* proto16   = (h16*)alloc((size_t)16 * 64 * 2);
  h16* pooledT[3];
  for (int i = 0; i < 3; ++i) pooledT[i] = (h16*)alloc((size_t)NROI * NS * NC * 2);
  h16* rcflat    = (h16*)alloc((size_t)NROI * KFC * 2);
  float* fcout   = (float*)alloc((size_t)NROI * 64 * 4);
  float* roi32   = (float*)alloc((size_t)NROI * 64 * 4);
  h16* resizedT  = (h16*)alloc((size_t)NB * 256 * 64 * 2);
  h16* keyT      = (h16*)alloc((size_t)NB * 256 * 64 * 2);
  h16* valT2     = (h16*)alloc((size_t)NB * 64 * 256 * 2);
  h16* queryh    = (h16*)alloc((size_t)NB * NP * 64 * 2);
  float* sim32   = (float*)alloc((size_t)NB * NP * 256 * 4);
  h16* simh      = (h16*)alloc((size_t)NB * NP * 256 * 2);
  h16* ffh       = (h16*)alloc((size_t)NROI * 64 * 2);
  h16* hbuf1     = (h16*)alloc((size_t)NROI * 64 * 2);
  h16* hbuf2     = (h16*)alloc((size_t)NROI * 64 * 2);
  float* clsl32  = (float*)alloc((size_t)NROI * 2 * 4);
  float* reg32   = (float*)alloc((size_t)NROI * 76 * 4);
  float* lines32 = (float*)alloc((size_t)NROI * 78 * 4);
  float* catf32  = (float*)alloc((size_t)NROI * 64 * 4);

  // Dispatch: full 64-wide strips with NSUB=4, plus one tail strip (NSUB 1-3).
  auto G = [&](const h16* A, int lda, long aBS, const h16* Bt, int ldb, long bBS,
               float* o32, h16* o16, int ldc, long oBS, int M, int N, int K,
               const float* cs, const float* cb, const float* rs, const float* rb,
               const float* add, long addBS, int addld, float alpha,
               int relu, int trans, int ffperm, int batch) {
    const int Npad = (N + 15) & ~15;
    const int nfull = Npad / 64;
    const int ntail = (Npad % 64) / 16;
    if (nfull > 0) {
      dim3 grid(M / 64, nfull, batch);
      gemm16<4><<<grid, 128, 0, stream>>>(A, lda, aBS, Bt, ldb, bBS, o32, o16, ldc, oBS,
                                          M, N, K, 0, cs, cb, rs, rb, add, addBS, addld,
                                          alpha, relu, trans, ffperm);
    }
    if (ntail > 0) {
      dim3 grid(M / 64, 1, batch);
      const int nbase = nfull * 64;
      switch (ntail) {
        case 1: gemm16<1><<<grid, 128, 0, stream>>>(A, lda, aBS, Bt, ldb, bBS, o32, o16, ldc, oBS,
                  M, N, K, nbase, cs, cb, rs, rb, add, addBS, addld, alpha, relu, trans, ffperm); break;
        case 2: gemm16<2><<<grid, 128, 0, stream>>>(A, lda, aBS, Bt, ldb, bBS, o32, o16, ldc, oBS,
                  M, N, K, nbase, cs, cb, rs, rb, add, addBS, addld, alpha, relu, trans, ffperm); break;
        default: gemm16<3><<<grid, 128, 0, stream>>>(A, lda, aBS, Bt, ldb, bBS, o32, o16, ldc, oBS,
                  M, N, K, nbase, cs, cb, rs, rb, add, addBS, addld, alpha, relu, trans, ffperm); break;
      }
    }
  };

  // ---- weight packing (recomputed each call; deterministic) ----
  {
    int n1 = 3 * 9 * 48 * 64;
    pack_rgw<<<(n1 + 255) / 256, 256, 0, stream>>>(rg_conv_w, rgw16);
    pack_catw<<<(9 * 64 * 64 + 255) / 256, 256, 0, stream>>>(catw0, catw16, 48, 64);
    pack_catw<<<(9 * 64 * 96 + 255) / 256, 256, 0, stream>>>(catw1, catw16 + 36864, 96, 96);
    pack_catw<<<(9 * 64 * 160 + 255) / 256, 256, 0, stream>>>(catw2, catw16 + 92160, 144, 160);
    auto CVT = [&](const float* s, h16* d, int r, int c, int rp) {
      cvt_pad_f16<<<(rp * c + 255) / 256, 256, 0, stream>>>(s, d, r, c, rp);
    };
    CVT(fc_w, fcw16, 64, KFC, 64);
    CVT(fkey_w, fkeyw16, 64, 64, 64);   CVT(fval_w, fvalw16, 64, 64, 64);
    CVT(cls_w1, clsw1_16, 64, 64, 64);  CVT(cls_w2, clsw2_16, 64, 64, 64);
    CVT(reg_w1, regw1_16, 64, 64, 64);  CVT(reg_w2, regw2_16, 64, 64, 64);
    CVT(cat_w1, catw1_16, 64, 64, 64);  CVT(cat_w2, catw2_16, 64, 64, 64);
    CVT(attr_w1, attrw1_16, 64, 64, 64); CVT(attr_w2, attrw2_16, 64, 64, 64);
    CVT(clsh_w, clshead16, 2, 64, 16);
    CVT(regh_w, reghead16, 76, 64, 80);
    CVT(attrh_w, attrhead16, 4, 64, 16);
    proto_norm<<<16, 64, 0, stream>>>(protos, proto16);
  }

  const float* featp[3] = {feat_s32, feat_s16, feat_s8};
  const int hh[3] = {10, 20, 40}, ww[3] = {25, 50, 100};
  const size_t catoff[3] = {0, 36864, 92160};

  for (int s = 0; s < 3; ++s) {
    const float* feat = featp[s];
    const int h = hh[s], w = ww[s];
    // 1) grid-sample pooling with this stage's priors
    pool_kernel<<<NROI, 256, 0, stream>>>(feat, h, w, lines32, s, pooledT[s]);
    // 2) fused conv stack (WMMA)
    switch (s) {
      case 0: conv_stack<0><<<NROI, 128, 0, stream>>>(pooledT[0], pooledT[1], pooledT[2],
                rgw16, rg_scale, rg_bias, catw16 + catoff[0], cat_scale + 0, cat_bias + 0, rcflat); break;
      case 1: conv_stack<1><<<NROI, 128, 0, stream>>>(pooledT[0], pooledT[1], pooledT[2],
                rgw16, rg_scale, rg_bias, catw16 + catoff[1], cat_scale + 64, cat_bias + 64, rcflat); break;
      default: conv_stack<2><<<NROI, 128, 0, stream>>>(pooledT[0], pooledT[1], pooledT[2],
                rgw16, rg_scale, rg_bias, catw16 + catoff[2], cat_scale + 128, cat_bias + 128, rcflat); break;
    }
    // 3) FC (6144x64x2304) + LayerNorm + relu
    G(rcflat, KFC, 0, fcw16, KFC, 0, fcout, nullptr, 64, 0, NROI, 64, KFC,
      nullptr, fc_b, nullptr, nullptr, nullptr, 0, 0, 1.f, 0, 0, 0, 1);
    ln_relu<<<NROI, 64, 0, stream>>>(fcout, roi32, ln_gamma, ln_beta);
    // 4) attention: resize -> key/value (batched), query, sim, softmax, context
    resize_kernel<<<(NB * 256 * 64 + 255) / 256, 256, 0, stream>>>(feat, h, w, resizedT);
    G(resizedT, 64, 256 * 64, fkeyw16, 64, 0, nullptr, keyT, 64, 256 * 64,
      256, 64, 64, fkey_s, fkey_b, nullptr, nullptr, nullptr, 0, 0, 1.f, 1, 0, 0, NB);
    G(resizedT, 64, 256 * 64, fvalw16, 64, 0, nullptr, valT2, 256, 64 * 256,
      256, 64, 64, nullptr, fval_b, nullptr, nullptr, nullptr, 0, 0, 1.f, 0, 1, 0, NB);
    query_kernel<<<(NROI * 64 + 255) / 256, 256, 0, stream>>>(roi32, fq_w, fq_b, queryh);
    G(queryh, 64, NP * 64, keyT, 64, 256 * 64, sim32, nullptr, 256, NP * 256,
      NP, 256, 64, nullptr, nullptr, nullptr, nullptr, nullptr, 0, 0, 1.f, 0, 0, 0, NB);
    softmax250<<<NROI, 256, 0, stream>>>(sim32, simh);
    // context + attn scale/bias + residual roi, fused ff row permutation -> ffh (f16)
    G(simh, 256, NP * 256, valT2, 256, 64 * 256, nullptr, ffh, 64, 0,
      NP, 64, 256, nullptr, nullptr, attn_s, attn_b, roi32, NP * 64, 64, 1.f, 0, 0, 1, NB);
    // 5) cls / reg heads
    G(ffh, 64, 0, clsw1_16, 64, 0, nullptr, hbuf1, 64, 0, NROI, 64, 64,
      nullptr, cls_b1, nullptr, nullptr, nullptr, 0, 0, 1.f, 1, 0, 0, 1);
    G(hbuf1, 64, 0, clsw2_16, 64, 0, nullptr, hbuf2, 64, 0, NROI, 64, 64,
      nullptr, cls_b2, nullptr, nullptr, nullptr, 0, 0, 1.f, 1, 0, 0, 1);
    G(hbuf2, 64, 0, clshead16, 64, 0, clsl32, nullptr, 2, 0, NROI, 2, 64,
      nullptr, clsh_b, nullptr, nullptr, nullptr, 0, 0, 1.f, 0, 0, 0, 1);
    G(ffh, 64, 0, regw1_16, 64, 0, nullptr, hbuf1, 64, 0, NROI, 64, 64,
      nullptr, reg_b1, nullptr, nullptr, nullptr, 0, 0, 1.f, 1, 0, 0, 1);
    G(hbuf1, 64, 0, regw2_16, 64, 0, nullptr, hbuf2, 64, 0, NROI, 64, 64,
      nullptr, reg_b2, nullptr, nullptr, nullptr, 0, 0, 1.f, 1, 0, 0, 1);
    G(hbuf2, 64, 0, reghead16, 64, 0, reg32, nullptr, 76, 0, NROI, 76, 64,
      nullptr, regh_b, nullptr, nullptr, nullptr, 0, 0, 1.f, 0, 0, 0, 1);
    // 6) lines / predictions (lines feed next stage's priors)
    lines_kernel<<<NROI, 128, 0, stream>>>(clsl32, reg32, lines32, dout, s);
  }

  // ---- category head (prototype cosine) ----
  G(ffh, 64, 0, catw1_16, 64, 0, nullptr, hbuf1, 64, 0, NROI, 64, 64,
    nullptr, cat_b1, nullptr, nullptr, nullptr, 0, 0, 1.f, 1, 0, 0, 1);
  G(hbuf1, 64, 0, catw2_16, 64, 0, catf32, nullptr, 64, 0, NROI, 64, 64,
    nullptr, cat_b2, nullptr, nullptr, nullptr, 0, 0, 1.f, 1, 0, 0, 1);
  rownorm64<<<NROI, 64, 0, stream>>>(catf32, hbuf2);  // reuse hbuf2 as catn16
  G(hbuf2, 64, 0, proto16, 64, 0, dout + (size_t)NROI * 78, nullptr, 15, 0,
    NROI, 15, 64, nullptr, nullptr, nullptr, nullptr, nullptr, 0, 0, 20.f, 0, 0, 0, 1);
  // ---- attribute head ----
  G(ffh, 64, 0, attrw1_16, 64, 0, nullptr, hbuf1, 64, 0, NROI, 64, 64,
    nullptr, attr_b1, nullptr, nullptr, nullptr, 0, 0, 1.f, 1, 0, 0, 1);
  G(hbuf1, 64, 0, attrw2_16, 64, 0, nullptr, hbuf2, 64, 0, NROI, 64, 64,
    nullptr, attr_b2, nullptr, nullptr, nullptr, 0, 0, 1.f, 1, 0, 0, 1);
  G(hbuf2, 64, 0, attrhead16, 64, 0, dout + (size_t)NROI * 93, nullptr, 4, 0,
    NROI, 4, 64, nullptr, attrh_b, nullptr, nullptr, nullptr, 0, 0, 1.f, 0, 0, 0, 1);
}
